// _GraphConvKerasModelExt_37417755083192
// MI455X (gfx1250) — matrix operand order repaired and checked by reading, not compiled, not run on hardware
//
#include <hip/hip_runtime.h>
#include <hip/hip_bf16.h>

// ---------------------------------------------------------------------------
// GraphConv model forward for MI455X (gfx1250, wave32, WMMA).
// GEMMs run on v_wmma_f32_16x16x32_bf16 (bf16 inputs, f32 accumulate).
// Weights are pre-packed into WMMA B-fragment order so every operand load
// is a pair of 16-byte vector loads (no strided u16 gather in the k-loop).
// ---------------------------------------------------------------------------

#define NATOMS   65536
#define NBATCH   1024
#define BN_EPS   1e-3f

// slab starts: [0, 4096, 16384, 36864, 57344], sizes [4096,12288,20480,20480,8192]

typedef __bf16 bf16_t;
typedef __attribute__((ext_vector_type(16))) __bf16 v16bf;
typedef __attribute__((ext_vector_type(8)))  __bf16 v8bf;
typedef __attribute__((ext_vector_type(8)))  float  v8f;

// ---------------------------------------------------------------------------
// helpers
// ---------------------------------------------------------------------------
__device__ __forceinline__ void atomic_max_f32(float* addr, float val) {
    unsigned int* ua = (unsigned int*)addr;
    unsigned int old = *ua;
    while (__uint_as_float(old) < val) {
        unsigned int assumed = old;
        old = atomicCAS(ua, assumed, __float_as_uint(val));
        if (old == assumed) break;
    }
}

__device__ __forceinline__ void slab_info(int atom, int& d, int& start) {
    if      (atom < 4096)  { d = 0; start = 0;     }
    else if (atom < 16384) { d = 1; start = 4096;  }
    else if (atom < 36864) { d = 2; start = 16384; }
    else if (atom < 57344) { d = 3; start = 36864; }
    else                   { d = 4; start = 57344; }
}

// ---------------------------------------------------------------------------
// weight convert + K-pad + pack into WMMA B-fragment order.
// Fragment (mat, kt, nt): 32 lanes x 16 contiguous bf16.
//   lane l, elem e  ->  W[kt*32 + 8*(l>>4) + e + (e>=8 ? 8 : 0)][nt*16 + (l&15)]
// ---------------------------------------------------------------------------
__global__ void pack_w_kernel(const float* __restrict__ src, bf16_t* __restrict__ dst,
                              int nmat, int Ksrc, int Ktiles, int Ntiles, int N) {
    int idx = blockIdx.x * blockDim.x + threadIdx.x;
    int total = nmat * Ktiles * Ntiles * 512;
    if (idx >= total) return;
    int e    = idx & 15;
    int lane = (idx >> 4) & 31;
    int t    = idx >> 9;                 // fragment index
    int nt   = t % Ntiles;
    int t2   = t / Ntiles;
    int kt   = t2 % Ktiles;
    int mat  = t2 / Ktiles;
    int k = kt * 32 + 8 * (lane >> 4) + e + (e >= 8 ? 8 : 0);
    int n = nt * 16 + (lane & 15);
    float v = (k < Ksrc) ? src[((long)mat * Ksrc + k) * N + n] : 0.0f;
    dst[idx] = (bf16_t)v;
}

// ---------------------------------------------------------------------------
// prep: f32 activations -> bf16 A_self (K-padded); optionally gather
// neighbor-sum into bf16 A_rel (degree slabs 1..4).
// ---------------------------------------------------------------------------
__global__ void prep_kernel(const float* __restrict__ src, int Ksrc, int Kpad,
                            bf16_t* __restrict__ Aself, bf16_t* __restrict__ Arel,
                            const int* __restrict__ adj1, const int* __restrict__ adj2,
                            const int* __restrict__ adj3, const int* __restrict__ adj4) {
    int idx = blockIdx.x * blockDim.x + threadIdx.x;
    int total = NATOMS * Kpad;
    if (idx >= total) return;
    int atom = idx / Kpad;
    int k    = idx - atom * Kpad;
    float sv = (k < Ksrc) ? src[(long)atom * Ksrc + k] : 0.0f;
    Aself[idx] = (bf16_t)sv;
    if (Arel && atom >= 4096) {
        float rv = 0.0f;
        if (k < Ksrc) {
            int d, start; slab_info(atom, d, start);
            const int* adj = (d == 1 ? adj1 : d == 2 ? adj2 : d == 3 ? adj3 : adj4)
                             + (long)(atom - start) * d;
            for (int j = 0; j < d; ++j)
                rv += src[(long)adj[j] * Ksrc + k];
        }
        Arel[idx] = (bf16_t)rv;
    }
}

// ---------------------------------------------------------------------------
// fragment loaders — both are two 16-byte vector loads per lane
// ---------------------------------------------------------------------------
__device__ __forceinline__ v16bf load_a_frag(const bf16_t* __restrict__ Arow,
                                             int k0, int khalf) {
    // lane (l&15 = row): elems 0..7 -> K = k0+8*khalf+e ; elems 8..15 -> +16
    const v8bf* p0 = (const v8bf*)(Arow + k0 + 8 * khalf);
    const v8bf* p1 = (const v8bf*)(Arow + k0 + 16 + 8 * khalf);
    v8bf lo = *p0, hi = *p1;
    v16bf a;
#pragma unroll
    for (int e = 0; e < 8; ++e) { a[e] = lo[e]; a[e + 8] = hi[e]; }
    return a;
}

__device__ __forceinline__ v16bf load_b_frag(const bf16_t* __restrict__ Wp,
                                             long mat, int kt, int nt,
                                             int Ktiles, int Ntiles, int lane) {
    long off = ((((mat * Ktiles + kt) * Ntiles + nt) * 32) + lane) * 16;
    return *(const v16bf*)(Wp + off);    // 32-byte aligned packed fragment
}

// ---------------------------------------------------------------------------
// graph-conv GEMM: per 16x16 C tile (one wave each):
//   slab0:  C = self @ W[8]
//   slab d: C = rel @ W[2d-2] + self @ W[2d-1]
// epilogue: +bias -> relu -> batchnorm -> f32 out
// ---------------------------------------------------------------------------
__global__ void gemm_conv_kernel(const bf16_t* __restrict__ Aself,
                                 const bf16_t* __restrict__ Arel,
                                 const bf16_t* __restrict__ Wp,    // packed, 9 mats
                                 const float*  __restrict__ bias,  // 9 x 256
                                 const float*  __restrict__ bn,    // 4 x 256
                                 float* __restrict__ out,          // NATOMS x 256
                                 int Ktiles) {
    const int NC = 256, NT = 16;
    int lane = threadIdx.x & 31;
    int wave = __builtin_amdgcn_readfirstlane(
        blockIdx.x * (blockDim.x >> 5) + (threadIdx.x >> 5));
    if (wave >= (NATOMS / 16) * NT) return;
    int mt = wave / NT, nt = wave - mt * NT;
    int m0 = mt * 16, n0 = nt * 16;
    int khalf = lane >> 4;
    int l15   = lane & 15;
    int Kpad  = Ktiles * 32;

    int slab, start; slab_info(m0, slab, start);
    int wself = (slab == 0) ? 8 : 2 * slab - 1;
    int wrel  = (slab == 0) ? 0 : 2 * slab - 2;

    const bf16_t* aS = Aself + (long)(m0 + l15) * Kpad;
    const bf16_t* aR = Arel  + (long)(m0 + l15) * Kpad;
    int col = n0 + l15;

    v8f c = {};
    for (int kt = 0; kt < Ktiles; ++kt) {
        int k0 = kt * 32;
        v16bf a = load_a_frag(aS, k0, khalf);
        v16bf b = load_b_frag(Wp, wself, kt, nt, Ktiles, NT, lane);
        c = __builtin_amdgcn_wmma_f32_16x16x32_bf16(false, a, false, b,
                                                    (short)0, c, false, false);
        if (slab > 0) {
            v16bf ar = load_a_frag(aR, k0, khalf);
            v16bf br = load_b_frag(Wp, wrel, kt, nt, Ktiles, NT, lane);
            c = __builtin_amdgcn_wmma_f32_16x16x32_bf16(false, ar, false, br,
                                                        (short)0, c, false, false);
        }
    }

    float bsum = bias[wself * NC + col];
    if (slab > 0) bsum += bias[wrel * NC + col];
    float g  = bn[0 * NC + col], bb = bn[1 * NC + col];
    float mm = bn[2 * NC + col], vv = bn[3 * NC + col];
    float inv = rsqrtf(vv + BN_EPS);
#pragma unroll
    for (int r = 0; r < 8; ++r) {
        int row = m0 + r + 8 * khalf;
        float y = c[r] + bsum;
        y = fmaxf(y, 0.0f);               // relu
        y = g * (y - mm) * inv + bb;      // batchnorm
        out[(long)row * NC + col] = y;
    }
}

// ---------------------------------------------------------------------------
// dense GEMM 256 -> 512: relu + bn3, bf16 output (feeds segment reductions)
// ---------------------------------------------------------------------------
__global__ void gemm_dense_kernel(const bf16_t* __restrict__ A,     // NATOMS x 256
                                  const bf16_t* __restrict__ Wp,    // packed 8x32 frags
                                  const float*  __restrict__ bias,  // 512
                                  const float*  __restrict__ bn,    // 4 x 512
                                  bf16_t* __restrict__ out) {       // NATOMS x 512
    const int K = 256, NC = 512, NT = 32, KT = 8;
    int lane = threadIdx.x & 31;
    int wave = __builtin_amdgcn_readfirstlane(
        blockIdx.x * (blockDim.x >> 5) + (threadIdx.x >> 5));
    if (wave >= (NATOMS / 16) * NT) return;
    int mt = wave / NT, nt = wave - mt * NT;
    int m0 = mt * 16, n0 = nt * 16;
    int khalf = lane >> 4;
    int l15   = lane & 15;
    const bf16_t* aRow = A + (long)(m0 + l15) * K;
    int col = n0 + l15;

    v8f c = {};
#pragma unroll
    for (int kt = 0; kt < KT; ++kt) {
        v16bf a = load_a_frag(aRow, kt * 32, khalf);
        v16bf b = load_b_frag(Wp, 0, kt, nt, KT, NT, lane);
        c = __builtin_amdgcn_wmma_f32_16x16x32_bf16(false, a, false, b,
                                                    (short)0, c, false, false);
    }
    float bsum = bias[col];
    float g  = bn[0 * NC + col], bb = bn[1 * NC + col];
    float mm = bn[2 * NC + col], vv = bn[3 * NC + col];
    float inv = rsqrtf(vv + BN_EPS);
#pragma unroll
    for (int r = 0; r < 8; ++r) {
        int row = m0 + r + 8 * khalf;
        float y = c[r] + bsum;
        y = fmaxf(y, 0.0f);
        y = g * (y - mm) * inv + bb;
        out[(long)row * NC + col] = (bf16_t)y;
    }
}

// ---------------------------------------------------------------------------
// graph pool: x_out = max(self, max over neighbors), identity for slab 0
// ---------------------------------------------------------------------------
__global__ void pool_kernel(const float* __restrict__ y, float* __restrict__ xout,
                            const int* __restrict__ adj1, const int* __restrict__ adj2,
                            const int* __restrict__ adj3, const int* __restrict__ adj4) {
    int idx = blockIdx.x * blockDim.x + threadIdx.x;
    if (idx >= NATOMS * 256) return;
    int atom = idx >> 8;
    int ch   = idx & 255;
    float v = y[idx];
    if (atom >= 4096) {
        int d, start; slab_info(atom, d, start);
        const int* adj = (d == 1 ? adj1 : d == 2 ? adj2 : d == 3 ? adj3 : adj4)
                         + (long)(atom - start) * d;
        for (int j = 0; j < d; ++j)
            v = fmaxf(v, y[(long)adj[j] * 256 + ch]);
    }
    xout[idx] = v;
}

// ---------------------------------------------------------------------------
// segment reductions
// ---------------------------------------------------------------------------
__global__ void seg_init_kernel(float* __restrict__ ssum, float* __restrict__ smax) {
    int idx = blockIdx.x * blockDim.x + threadIdx.x;
    if (idx >= NBATCH * 512) return;
    ssum[idx] = 0.0f;
    smax[idx] = -__builtin_inff();
}

__global__ void seg_acc_kernel(const bf16_t* __restrict__ h, const int* __restrict__ memb,
                               float* __restrict__ ssum, float* __restrict__ smax) {
    int idx = blockIdx.x * blockDim.x + threadIdx.x;
    if (idx >= NATOMS * 512) return;
    int atom = idx >> 9;
    int ch   = idx & 511;
    float v = (float)h[idx];
    int s = memb[atom];
    atomicAdd(&ssum[(long)s * 512 + ch], v);
    atomic_max_f32(&smax[(long)s * 512 + ch], v);
}

// fp = tanh(concat(seg_sum, seg_max))  ->  d_out fp region
__global__ void fp_kernel(const float* __restrict__ ssum, const float* __restrict__ smax,
                          float* __restrict__ fp) {
    int idx = blockIdx.x * blockDim.x + threadIdx.x;
    if (idx >= NBATCH * 1024) return;
    int s  = idx >> 10;
    int ch = idx & 1023;
    float v = (ch < 512) ? ssum[(long)s * 512 + ch] : smax[(long)s * 512 + (ch - 512)];
    fp[idx] = tanhf(v);
}

// logits = fp @ head_W + head_b      (1024 x 1024) @ (1024 x 24)
__global__ void head_kernel(const float* __restrict__ fp, const float* __restrict__ hW,
                            const float* __restrict__ hb, float* __restrict__ logits) {
    int idx = blockIdx.x * blockDim.x + threadIdx.x;
    if (idx >= NBATCH * 24) return;
    int s = idx / 24, cc = idx - s * 24;
    const float* f = fp + (long)s * 1024;
    float acc = hb[cc];
    for (int k = 0; k < 1024; ++k)
        acc += f[k] * hW[k * 24 + cc];
    logits[idx] = acc;
}

// softmax over the 2-class axis
__global__ void softmax_kernel(const float* __restrict__ logits, float* __restrict__ sm) {
    int idx = blockIdx.x * blockDim.x + threadIdx.x;
    if (idx >= NBATCH * 12) return;
    float l0 = logits[idx * 2 + 0];
    float l1 = logits[idx * 2 + 1];
    float m  = fmaxf(l0, l1);
    float e0 = __expf(l0 - m);
    float e1 = __expf(l1 - m);
    float inv = 1.0f / (e0 + e1);
    sm[idx * 2 + 0] = e0 * inv;
    sm[idx * 2 + 1] = e1 * inv;
}

// ---------------------------------------------------------------------------
// launch
// ---------------------------------------------------------------------------
extern "C" void kernel_launch(void* const* d_in, const int* in_sizes, int n_in,
                              void* d_out, int out_size, void* d_ws, size_t ws_size,
                              hipStream_t stream) {
    (void)in_sizes; (void)n_in; (void)out_size; (void)ws_size;

    const float* atom_feat = (const float*)d_in[0];
    const int*   membership= (const int*)  d_in[2];
    const int*   adj1      = (const int*)  d_in[4];
    const int*   adj2      = (const int*)  d_in[5];
    const int*   adj3      = (const int*)  d_in[6];
    const int*   adj4      = (const int*)  d_in[7];
    const float* gc1_W     = (const float*)d_in[8];
    const float* gc1_b     = (const float*)d_in[9];
    const float* gc2_W     = (const float*)d_in[10];
    const float* gc2_b     = (const float*)d_in[11];
    const float* bn1       = (const float*)d_in[12];
    const float* bn2       = (const float*)d_in[13];
    const float* bn3       = (const float*)d_in[14];
    const float* dense_W   = (const float*)d_in[15];
    const float* dense_b   = (const float*)d_in[16];
    const float* head_W    = (const float*)d_in[17];
    const float* head_b    = (const float*)d_in[18];

    float* out_sm     = (float*)d_out;              // 1024*12*2
    float* out_logits = out_sm + NBATCH * 24;       // 1024*12*2
    float* out_fp     = out_logits + NBATCH * 24;   // 1024*1024

    // scratch layout
    char*  base = (char*)d_ws;
    size_t off  = 0;
    auto  alloc = [&](size_t bytes) -> void* {
        void* p = base + off;
        off = (off + bytes + 255) & ~(size_t)255;
        return p;
    };
    bf16_t* wsW1   = (bf16_t*)alloc((size_t)9 * 3 * 16 * 512 * 2);  // gc1 packed (K 75->96)
    bf16_t* wsW2   = (bf16_t*)alloc((size_t)9 * 8 * 16 * 512 * 2);  // gc2 packed
    bf16_t* wsWd   = (bf16_t*)alloc((size_t)1 * 8 * 32 * 512 * 2);  // dense packed
    bf16_t* Aself  = (bf16_t*)alloc((size_t)NATOMS * 256 * 2);
    bf16_t* Arel   = (bf16_t*)alloc((size_t)NATOMS * 256 * 2);
    float*  bufY   = (float*) alloc((size_t)NATOMS * 256 * 4);      // conv out; reused as bf16 h
    float*  bufX   = (float*) alloc((size_t)NATOMS * 256 * 4);      // pooled activations
    float*  segSum = (float*) alloc((size_t)NBATCH * 512 * 4);
    float*  segMax = (float*) alloc((size_t)NBATCH * 512 * 4);
    bf16_t* hBf    = (bf16_t*)bufY;                                 // dense output, 65536 x 512 bf16

    const int TPB = 256;
    auto blocks = [](long n, int tpb) { return (unsigned)((n + tpb - 1) / tpb); };

    // 0) weight conversion: bf16, K-padded, packed into WMMA fragments
    pack_w_kernel<<<blocks((long)9 * 3 * 16 * 512, TPB), TPB, 0, stream>>>(gc1_W, wsW1, 9, 75, 3, 16, 256);
    pack_w_kernel<<<blocks((long)9 * 8 * 16 * 512, TPB), TPB, 0, stream>>>(gc2_W, wsW2, 9, 256, 8, 16, 256);
    pack_w_kernel<<<blocks((long)1 * 8 * 32 * 512, TPB), TPB, 0, stream>>>(dense_W, wsWd, 1, 256, 8, 32, 512);

    // ---- layer 1: conv(75->256) + bn1 + pool ----
    prep_kernel<<<blocks((long)NATOMS * 96, TPB), TPB, 0, stream>>>(
        atom_feat, 75, 96, Aself, Arel, adj1, adj2, adj3, adj4);
    gemm_conv_kernel<<<blocks((long)(NATOMS / 16) * 16 * 32, TPB), TPB, 0, stream>>>(
        Aself, Arel, wsW1, gc1_b, bn1, bufY, 3);
    pool_kernel<<<blocks((long)NATOMS * 256, TPB), TPB, 0, stream>>>(
        bufY, bufX, adj1, adj2, adj3, adj4);

    // ---- layer 2: conv(256->256) + bn2 + pool ----
    prep_kernel<<<blocks((long)NATOMS * 256, TPB), TPB, 0, stream>>>(
        bufX, 256, 256, Aself, Arel, adj1, adj2, adj3, adj4);
    gemm_conv_kernel<<<blocks((long)(NATOMS / 16) * 16 * 32, TPB), TPB, 0, stream>>>(
        Aself, Arel, wsW2, gc2_b, bn2, bufY, 8);
    pool_kernel<<<blocks((long)NATOMS * 256, TPB), TPB, 0, stream>>>(
        bufY, bufX, adj1, adj2, adj3, adj4);

    // ---- dense 256->512 + relu + bn3 (bf16 out) ----
    prep_kernel<<<blocks((long)NATOMS * 256, TPB), TPB, 0, stream>>>(
        bufX, 256, 256, Aself, nullptr, adj1, adj2, adj3, adj4);
    gemm_dense_kernel<<<blocks((long)(NATOMS / 16) * 32 * 32, TPB), TPB, 0, stream>>>(
        Aself, wsWd, dense_b, bn3, hBf);

    // ---- segment sum/max + tanh fingerprint ----
    seg_init_kernel<<<blocks((long)NBATCH * 512, TPB), TPB, 0, stream>>>(segSum, segMax);
    seg_acc_kernel<<<blocks((long)NATOMS * 512, TPB), TPB, 0, stream>>>(hBf, membership, segSum, segMax);
    fp_kernel<<<blocks((long)NBATCH * 1024, TPB), TPB, 0, stream>>>(segSum, segMax, out_fp);

    // ---- head + softmax ----
    head_kernel<<<blocks((long)NBATCH * 24, TPB), TPB, 0, stream>>>(out_fp, head_W, head_b, out_logits);
    softmax_kernel<<<blocks((long)NBATCH * 12, TPB), TPB, 0, stream>>>(out_logits, out_sm);
}
